// Attention_85985245266248
// MI455X (gfx1250) — matrix operand order repaired and checked by reading, not compiled, not run on hardware
//
#include <hip/hip_runtime.h>

// ---------------------------------------------------------------------------
// CDNA5 (gfx1250) dual-branch channel attention block.
// All dense math on v_wmma_f32_16x16x32_bf16 (wave32 WMMA), fp32 accumulate.
// LDS tiles are staged in WMMA *fragment order* so every fragment load is two
// ds_load_b128 (32 contiguous bytes per lane) with zero packing VALU.
// Memory-bound => bf16 intermediates; softmax/projection tail folded into one
// per-batch 96x192 matrix so the big V tensors are streamed exactly once.
// ---------------------------------------------------------------------------

#define NPIX 65536   // 256*256
#define IMG  256
#define BATCH 2

typedef __attribute__((ext_vector_type(16))) __bf16 v16bf;
typedef __attribute__((ext_vector_type(8)))  float  v8f;

// Native f32 -> bf16 conversion (lets the backend use v_cvt_* ops instead of
// a manual round-to-nearest-even bit sequence).
__device__ __forceinline__ unsigned short f2bf(float f) {
  union { __bf16 b; unsigned short u; } cv;
  cv.b = (__bf16)f;
  return cv.u;
}
__device__ __forceinline__ float bf2f(unsigned short s) {
  return __uint_as_float(((unsigned int)s) << 16);
}
// Pack two f32 as two bf16 in one dword (4B-aligned LDS store).
__device__ __forceinline__ unsigned int packcvt(float lo, float hi) {
  return (unsigned int)f2bf(lo) | ((unsigned int)f2bf(hi) << 16);
}

// WMMA 16-bit A (16x32) fragment coordinates: lane(m,k), element(k)
__device__ __forceinline__ int a_lane(int m, int k) { return m + 16 * ((k >> 3) & 1); }
__device__ __forceinline__ int a_elem(int k)        { return (k & 7) | ((k >> 4) << 3); }
// WMMA 16-bit B (32x16) fragment coordinates: lane(n,k), element(k)
__device__ __forceinline__ int b_lane(int n, int k) { return n + 16 * ((k >> 3) & 1); }
__device__ __forceinline__ int b_elem(int k)        { return (k & 7) | ((k >> 4) << 3); }

__device__ __forceinline__ v16bf frag(const unsigned short* s, int tile, int lane) {
  return *(const v16bf*)(s + ((size_t)(tile * 32 + lane)) * 16);
}

// ---------------------------------------------------------------------------
// Kernel 1: 1x1 conv as GEMM  Y[b,o,n] = sum_c W[o,c] * X[b,c,n]   (C = 96)
// block = 128 (4 waves); tile M=16, N=64, K=96 (3 k-tiles). Frag-order LDS.
// ---------------------------------------------------------------------------
__global__ void __launch_bounds__(128) conv1x1_wmma(
    const float* __restrict__ X,      // [B, 96, N] fp32
    const float* __restrict__ W,      // [O, 96]    fp32
    unsigned short* __restrict__ Y,   // [B, O, N]  bf16
    int O, int N)
{
  __shared__ __align__(32) unsigned short sW[3 * 32 * 16];       // kt tiles
  __shared__ __align__(32) unsigned short sX[12 * 32 * 16];      // (kt,nt) tiles
  const int tid  = threadIdx.x;
  const int lane = tid & 31;
  const int wv   = tid >> 5;
  const int half = lane >> 4;
  const int lr   = lane & 15;
  const int n0 = blockIdx.x * 64;
  const int m0 = blockIdx.y * 16;
  const int b  = blockIdx.z;
  const float* Xb = X + (size_t)b * 96 * N;

  // Stage W tile (16 x 96) in A-fragment order; pair along k (e, e+1).
  for (int i = tid; i < 16 * 48; i += 128) {
    int m = i / 48, k = (i % 48) * 2;
    const float* wp = W + (size_t)(m0 + m) * 96 + k;
    int kt = k >> 5, kl = k & 31;
    int idx = ((kt * 32 + a_lane(m, kl)) * 16) + a_elem(kl);
    *(unsigned int*)(sW + idx) = packcvt(wp[0], wp[1]);
  }
  // Stage X tile (96 x 64) in B-fragment order; pair along c (e, e+1).
  for (int i = tid; i < 48 * 64; i += 128) {
    int c = (i >> 6) * 2, p = i & 63;
    float x0 = Xb[(size_t)c * N + n0 + p];
    float x1 = Xb[(size_t)(c + 1) * N + n0 + p];
    int kt = c >> 5, cl = c & 31, nt = p >> 4;
    int idx = (((kt * 4 + nt) * 32 + b_lane(p & 15, cl)) * 16) + b_elem(cl);
    *(unsigned int*)(sX + idx) = packcvt(x0, x1);
  }
  __syncthreads();

  v8f acc = {};
#pragma unroll
  for (int kt = 0; kt < 3; ++kt) {
    v16bf a  = frag(sW, kt, lane);
    v16bf bm = frag(sX, kt * 4 + wv, lane);
    acc = __builtin_amdgcn_wmma_f32_16x16x32_bf16(false, a, false, bm,
                                                  (short)0, acc, false, false);
  }
#pragma unroll
  for (int r = 0; r < 8; ++r) {
    int m = m0 + r + half * 8;
    int n = n0 + wv * 16 + lr;
    Y[((size_t)b * O + m) * N + n] = f2bf(acc[r]);
  }
}

// ---------------------------------------------------------------------------
// Kernel 2: depthwise 3x3, SAME padding (cross-correlation), bf16 in/out.
// Fast path for interior pixels (no per-tap bounds checks).
// ---------------------------------------------------------------------------
__global__ void __launch_bounds__(256) dwconv3_kernel(
    const unsigned short* __restrict__ X,  // [B, C, 256, 256] bf16
    const float* __restrict__ Wd,          // [C, 9] fp32
    unsigned short* __restrict__ Y,        // [B, C, 256, 256] bf16
    int C)
{
  const int p  = blockIdx.x * 256 + threadIdx.x;
  const int ch = blockIdx.y;
  const int b  = blockIdx.z;
  const int x = p & 255, y = p >> 8;
  const unsigned short* Xc = X + ((size_t)b * C + ch) * NPIX;
  float w[9];
#pragma unroll
  for (int i = 0; i < 9; ++i) w[i] = Wd[ch * 9 + i];
  float acc = 0.f;
  if (x >= 1 && x <= 254 && y >= 1 && y <= 254) {
    const unsigned short* c0 = Xc + (y - 1) * IMG + (x - 1);
#pragma unroll
    for (int dy = 0; dy < 3; ++dy)
#pragma unroll
      for (int dx = 0; dx < 3; ++dx)
        acc += w[dy * 3 + dx] * bf2f(c0[dy * IMG + dx]);
  } else {
#pragma unroll
    for (int dy = -1; dy <= 1; ++dy) {
      int yy = y + dy;
      if (yy < 0 || yy >= IMG) continue;
#pragma unroll
      for (int dx = -1; dx <= 1; ++dx) {
        int xx = x + dx;
        if (xx < 0 || xx >= IMG) continue;
        acc += w[(dy + 1) * 3 + (dx + 1)] * bf2f(Xc[yy * IMG + xx]);
      }
    }
  }
  Y[((size_t)b * C + ch) * NPIX + p] = f2bf(acc);
}

// ---------------------------------------------------------------------------
// Kernel 3: per-channel sum of squares over n (for L2 normalization)
// ---------------------------------------------------------------------------
__global__ void __launch_bounds__(256) sumsq_kernel(
    const unsigned short* __restrict__ X, float* __restrict__ out,
    int CS, int N)
{
  const int ch = blockIdx.x, b = blockIdx.y;
  const uint4* Xc = (const uint4*)(X + ((size_t)b * CS + ch) * (size_t)N);
  float s = 0.f;
  for (int i = threadIdx.x; i < N / 8; i += 256) {
    uint4 v = Xc[i];
#pragma unroll
    for (int j = 0; j < 4; ++j) {
      unsigned int d = (&v.x)[j];
      float a0 = bf2f((unsigned short)(d & 0xffffu));
      float a1 = bf2f((unsigned short)(d >> 16));
      s += a0 * a0 + a1 * a1;
    }
  }
  __shared__ float red[256];
  red[threadIdx.x] = s;
  __syncthreads();
  for (int st = 128; st > 0; st >>= 1) {
    if (threadIdx.x < st) red[threadIdx.x] += red[threadIdx.x + st];
    __syncthreads();
  }
  if (threadIdx.x == 0) out[b * 96 + ch] = red[0];
}

// ---------------------------------------------------------------------------
// Kernel 4: Gram (Q.K^T over n=65536) via WMMA + normalize + softmax.
// One workgroup per (head, batch); 8 waves each own a 32-pixel K-slice per
// iteration; LDS tiles staged in fragment order (u32 copies: already bf16).
// ---------------------------------------------------------------------------
__global__ void __launch_bounds__(256) gram_softmax_wmma(
    const unsigned short* __restrict__ Q,   // [B, 96, N]  bf16
    const unsigned short* __restrict__ Kb,  // [B, 192, N] bf16 (k = ch 0..95)
    const float* __restrict__ ssQ,          // [B*96]
    const float* __restrict__ ssK,          // [B*96]
    const float* __restrict__ temp,         // [3]
    float* __restrict__ attn,               // [B, 3, 32, 32]
    int N)
{
  __shared__ __align__(32) unsigned short sQ[16 * 32 * 16];  // (wv, mt) tiles
  __shared__ __align__(32) unsigned short sK[16 * 32 * 16];  // (wv, nt) tiles
  __shared__ float sG[32 * 32];
  const int h = blockIdx.x, b = blockIdx.y;
  const int tid = threadIdx.x, lane = tid & 31, wv = tid >> 5;
  const int half = lane >> 4, lr = lane & 15;
  const unsigned short* Qh = Q  + ((size_t)b * 96  + h * 32) * (size_t)N;
  const unsigned short* Kh = Kb + ((size_t)b * 192 + h * 32) * (size_t)N;

  v8f g00 = {}, g01 = {}, g10 = {}, g11 = {};
  for (int n0 = 0; n0 < N; n0 += 256) {
    __syncthreads();
    // Stage Q in A-frag order, K in B-frag order; pixels paired (u32 copies).
    for (int i = tid; i < 32 * 128; i += 256) {
      int c = i >> 7, p = (i & 127) * 2;
      unsigned int qv = *(const unsigned int*)(Qh + (size_t)c * N + n0 + p);
      unsigned int kv = *(const unsigned int*)(Kh + (size_t)c * N + n0 + p);
      int w = p >> 5, pl = p & 31;
      int qidx = (((w * 2 + (c >> 4)) * 32 + a_lane(c & 15, pl)) * 16) + a_elem(pl);
      int kidx = (((w * 2 + (c >> 4)) * 32 + b_lane(c & 15, pl)) * 16) + b_elem(pl);
      *(unsigned int*)(sQ + qidx) = qv;
      *(unsigned int*)(sK + kidx) = kv;
    }
    __syncthreads();
    v16bf a0 = frag(sQ, wv * 2 + 0, lane);
    v16bf a1 = frag(sQ, wv * 2 + 1, lane);
    v16bf b0 = frag(sK, wv * 2 + 0, lane);
    v16bf b1 = frag(sK, wv * 2 + 1, lane);
    g00 = __builtin_amdgcn_wmma_f32_16x16x32_bf16(false, a0, false, b0, (short)0, g00, false, false);
    g01 = __builtin_amdgcn_wmma_f32_16x16x32_bf16(false, a0, false, b1, (short)0, g01, false, false);
    g10 = __builtin_amdgcn_wmma_f32_16x16x32_bf16(false, a1, false, b0, (short)0, g10, false, false);
    g11 = __builtin_amdgcn_wmma_f32_16x16x32_bf16(false, a1, false, b1, (short)0, g11, false, false);
  }

  __syncthreads();
  for (int i = tid; i < 32 * 32; i += 256) sG[i] = 0.f;
  __syncthreads();
#pragma unroll
  for (int r = 0; r < 8; ++r) {
    int row = r + half * 8, col = lr;
    atomicAdd(&sG[(row     ) * 32 + col     ], g00[r]);
    atomicAdd(&sG[(row     ) * 32 + col + 16], g01[r]);
    atomicAdd(&sG[(row + 16) * 32 + col     ], g10[r]);
    atomicAdd(&sG[(row + 16) * 32 + col + 16], g11[r]);
  }
  __syncthreads();

  if (tid < 32) {
    const int c = tid;
    const float t  = temp[h];
    const float iq = 1.f / fmaxf(sqrtf(ssQ[b * 96 + h * 32 + c]), 1e-12f);
    float row[32], mx = -3.4e38f;
#pragma unroll
    for (int d = 0; d < 32; ++d) {
      float ik = 1.f / fmaxf(sqrtf(ssK[b * 96 + h * 32 + d]), 1e-12f);
      row[d] = sG[c * 32 + d] * iq * ik * t;
      mx = fmaxf(mx, row[d]);
    }
    float sum = 0.f;
#pragma unroll
    for (int d = 0; d < 32; ++d) { row[d] = expf(row[d] - mx); sum += row[d]; }
    float inv = 1.f / sum;
    float* ap = attn + (((size_t)b * 3 + h) * 32 + c) * 32;
#pragma unroll
    for (int d = 0; d < 32; ++d) ap[d] = row[d] * inv;
  }
}

// ---------------------------------------------------------------------------
// Kernel 5a: Mc = concat_w[:, :96] @ po_c ; Mt = concat_w[:, 96:] @ po_t
// ---------------------------------------------------------------------------
__global__ void __launch_bounds__(256) mergeM_kernel(
    const float* __restrict__ cw, const float* __restrict__ poc,
    const float* __restrict__ pot, float* __restrict__ Mc, float* __restrict__ Mt)
{
  int idx = blockIdx.x * 256 + threadIdx.x;
  if (idx >= 96 * 96) return;
  int o = idx / 96, c = idx % 96;
  float sc = 0.f, st = 0.f;
  for (int j = 0; j < 96; ++j) {
    sc += cw[o * 192 + j]      * poc[j * 96 + c];
    st += cw[o * 192 + 96 + j] * pot[j * 96 + c];
  }
  Mc[idx] = sc; Mt[idx] = st;
}

// ---------------------------------------------------------------------------
// Kernel 5b: P[b] = [ Mc @ blockdiag(attn_c[b]) | Mt @ blockdiag(attn_t[b]) ]
// ---------------------------------------------------------------------------
__global__ void __launch_bounds__(256) buildP_kernel(
    const float* __restrict__ Mc, const float* __restrict__ Mt,
    const float* __restrict__ ac, const float* __restrict__ at,
    unsigned short* __restrict__ P)
{
  int idx = blockIdx.x * 256 + threadIdx.x;
  if (idx >= BATCH * 96 * 192) return;
  int b = idx / (96 * 192);
  int rem = idx % (96 * 192);
  int o = rem / 192, k = rem % 192;
  int s = k / 96, kk = k % 96;
  int h = kk / 32, d = kk % 32;
  const float* M = s ? Mt : Mc;
  const float* A = s ? at : ac;
  float acc = 0.f;
  for (int c = 0; c < 32; ++c)
    acc += M[o * 96 + h * 32 + c] * A[(((size_t)b * 3 + h) * 32 + c) * 32 + d];
  P[idx] = f2bf(acc);
}

// ---------------------------------------------------------------------------
// Kernel 6: final fused GEMM  out[b,o,n] = sum_k P[b,o,k] * V[k,n] + bias[o]
// V rows 0..95 = kv_t v-part; rows 96..191 = kv_c v-part. M=96, K=192.
// ---------------------------------------------------------------------------
__global__ void __launch_bounds__(128) final_gemm_wmma(
    const unsigned short* __restrict__ P,    // [B, 96, 192] bf16
    const unsigned short* __restrict__ KVt,  // [B, 192, N]  bf16
    const unsigned short* __restrict__ KVc,  // [B, 192, N]  bf16
    const float* __restrict__ bias,          // [96]
    float* __restrict__ out,                 // [B, 96, N]   fp32
    int N)
{
  __shared__ __align__(32) unsigned short sP[6 * 32 * 16];       // kt tiles
  __shared__ __align__(32) unsigned short sV[24 * 32 * 16];      // (kt,nt) tiles
  const int tid  = threadIdx.x;
  const int lane = tid & 31;
  const int wv   = tid >> 5;
  const int half = lane >> 4;
  const int lr   = lane & 15;
  const int n0 = blockIdx.x * 64;
  const int m0 = blockIdx.y * 16;
  const int b  = blockIdx.z;

  // Stage P tile (16 x 192) in A-frag order; k paired (u32 copy, already bf16)
  for (int i = tid; i < 16 * 96; i += 128) {
    int m = i / 96, k = (i % 96) * 2;
    unsigned int pv = *(const unsigned int*)(P + (size_t)b * 96 * 192 +
                                             (size_t)(m0 + m) * 192 + k);
    int kt = k >> 5, kl = k & 31;
    int idx = ((kt * 32 + a_lane(m, kl)) * 16) + a_elem(kl);
    *(unsigned int*)(sP + idx) = pv;
  }
  // Stage V tile (192 x 64) in B-frag order; pixels paired on the load side.
  for (int i = tid; i < 192 * 32; i += 128) {
    int k = i >> 5, p = (i & 31) * 2;
    const unsigned short* src = (k < 96)
        ? (KVt + ((size_t)b * 192 + 96 + k) * (size_t)N)
        : (KVc + ((size_t)b * 192 + 96 + (k - 96)) * (size_t)N);
    unsigned int vv = *(const unsigned int*)(src + n0 + p);
    int kt = k >> 5, kl = k & 31;
    int i0 = (((kt * 4 + (p >> 4)) * 32 + b_lane(p & 15, kl)) * 16) + b_elem(kl);
    int p1 = p + 1;
    int i1 = (((kt * 4 + (p1 >> 4)) * 32 + b_lane(p1 & 15, kl)) * 16) + b_elem(kl);
    sV[i0] = (unsigned short)(vv & 0xffffu);
    sV[i1] = (unsigned short)(vv >> 16);
  }
  __syncthreads();

  v8f acc = {};
#pragma unroll
  for (int kt = 0; kt < 6; ++kt) {
    v16bf a  = frag(sP, kt, lane);
    v16bf bm = frag(sV, kt * 4 + wv, lane);
    acc = __builtin_amdgcn_wmma_f32_16x16x32_bf16(false, a, false, bm,
                                                  (short)0, acc, false, false);
  }
#pragma unroll
  for (int r = 0; r < 8; ++r) {
    int m = m0 + r + half * 8;
    int n = n0 + wv * 16 + lr;
    out[((size_t)b * 96 + m) * N + n] = acc[r] + bias[m];
  }
}

// ---------------------------------------------------------------------------
extern "C" void kernel_launch(void* const* d_in, const int* in_sizes, int n_in,
                              void* d_out, int out_size, void* d_ws, size_t ws_size,
                              hipStream_t stream) {
  (void)in_sizes; (void)n_in; (void)out_size; (void)ws_size;
  const float* low    = (const float*)d_in[0];
  const float* high   = (const float*)d_in[1];
  const float* temp   = (const float*)d_in[2];
  const float* qc_w   = (const float*)d_in[3];
  const float* qdw_c  = (const float*)d_in[4];
  const float* kvc_w  = (const float*)d_in[5];
  const float* kvdw_c = (const float*)d_in[6];
  const float* qt_w   = (const float*)d_in[7];
  const float* qdw_t  = (const float*)d_in[8];
  const float* kvt_w  = (const float*)d_in[9];
  const float* kvdw_t = (const float*)d_in[10];
  const float* po_c   = (const float*)d_in[11];
  const float* po_t   = (const float*)d_in[12];
  const float* cat_w  = (const float*)d_in[13];
  const float* cat_b  = (const float*)d_in[14];
  float* out = (float*)d_out;

  const int N = NPIX, B = BATCH;
  char* base = (char*)d_ws;
  size_t off = 0;
  auto wsa = [&](size_t bytes) { char* p = base + off; off += (bytes + 255) & ~(size_t)255; return p; };

  unsigned short* pre   = (unsigned short*)wsa((size_t)B * 192 * N * 2); // reused scratch
  unsigned short* q_c   = (unsigned short*)wsa((size_t)B * 96  * N * 2);
  unsigned short* kv_c  = (unsigned short*)wsa((size_t)B * 192 * N * 2);
  unsigned short* q_t   = (unsigned short*)wsa((size_t)B * 96  * N * 2);
  unsigned short* kv_t  = (unsigned short*)wsa((size_t)B * 192 * N * 2);
  float* ss_qc = (float*)wsa((size_t)B * 96 * 4);
  float* ss_qt = (float*)wsa((size_t)B * 96 * 4);
  float* ss_kc = (float*)wsa((size_t)B * 96 * 4);
  float* ss_kt = (float*)wsa((size_t)B * 96 * 4);
  float* attn_c = (float*)wsa((size_t)B * 3 * 32 * 32 * 4);
  float* attn_t = (float*)wsa((size_t)B * 3 * 32 * 32 * 4);
  float* Mc = (float*)wsa(96 * 96 * 4);
  float* Mt = (float*)wsa(96 * 96 * 4);
  unsigned short* P = (unsigned short*)wsa((size_t)B * 96 * 192 * 2);

  const dim3 gConvQ(N / 64, 6, B), gConvKV(N / 64, 12, B);

  // Branch convs: conv1x1 (WMMA) -> depthwise 3x3, ping-pong through `pre`.
  conv1x1_wmma<<<gConvQ, 128, 0, stream>>>(high, qc_w, pre, 96, N);
  dwconv3_kernel<<<dim3(256, 96, B), 256, 0, stream>>>(pre, qdw_c, q_c, 96);
  conv1x1_wmma<<<gConvKV, 128, 0, stream>>>(high, kvc_w, pre, 192, N);
  dwconv3_kernel<<<dim3(256, 192, B), 256, 0, stream>>>(pre, kvdw_c, kv_c, 192);
  conv1x1_wmma<<<gConvQ, 128, 0, stream>>>(low, qt_w, pre, 96, N);
  dwconv3_kernel<<<dim3(256, 96, B), 256, 0, stream>>>(pre, qdw_t, q_t, 96);
  conv1x1_wmma<<<gConvKV, 128, 0, stream>>>(low, kvt_w, pre, 192, N);
  dwconv3_kernel<<<dim3(256, 192, B), 256, 0, stream>>>(pre, kvdw_t, kv_t, 192);

  // L2 norms (sum of squares along n).
  sumsq_kernel<<<dim3(96, B), 256, 0, stream>>>(q_c, ss_qc, 96, N);
  sumsq_kernel<<<dim3(96, B), 256, 0, stream>>>(q_t, ss_qt, 96, N);
  sumsq_kernel<<<dim3(96, B), 256, 0, stream>>>(kv_c, ss_kc, 192, N);
  sumsq_kernel<<<dim3(96, B), 256, 0, stream>>>(kv_t, ss_kt, 192, N);

  // Gram + softmax (cross-branch: q_c with k_t, q_t with k_c).
  gram_softmax_wmma<<<dim3(3, B), 256, 0, stream>>>(q_c, kv_t, ss_qc, ss_kt, temp, attn_c, N);
  gram_softmax_wmma<<<dim3(3, B), 256, 0, stream>>>(q_t, kv_c, ss_qt, ss_kc, temp, attn_t, N);

  // Fold projection chain into one per-batch 96x192 matrix.
  mergeM_kernel<<<dim3((96 * 96 + 255) / 256), 256, 0, stream>>>(cat_w, po_c, po_t, Mc, Mt);
  buildP_kernel<<<dim3((B * 96 * 192 + 255) / 256), 256, 0, stream>>>(Mc, Mt, attn_c, attn_t, P);

  // Final fused GEMM: streams v_t and v_c once, writes fp32 output + bias.
  final_gemm_wmma<<<dim3(N / 64, 6, B), 128, 0, stream>>>(P, kv_t, kv_c, cat_b, out, N);
}
